// MOERouter_30657476559564
// MI455X (gfx1250) — matrix operand order repaired and checked by reading, not compiled, not run on hardware
//
#include <hip/hip_runtime.h>

#define HIDDEN 768
#define INTER  3072
#define NE     8
#define NTOK   2048   // 2 * 1024
#define NGROUP 16     // NE experts x TOP_K slots
#define CAP    2048   // max tokens per (expert, k-slot) list

// ---- workspace layout ----
#define OFF_TOK  32                         // int units
#define OFF_WT   32800                      // float units
#define OFF_H1_BYTES  262272                            // 4096 x 3072 bf16 (~24 MB)
#define OFF_W1P_BYTES (OFF_H1_BYTES + 4096 * INTER * 2) // 25428096, 8x768x3072 bf16 (~36 MB)
#define OFF_W2P_BYTES (OFF_W1P_BYTES + NE * HIDDEN * INTER * 2) // + ~36 MB  (total ~96 MB)

// ---- d_out layout (floats, reference tuple flattened) ----
#define OUT_IDX   (NTOK * HIDDEN)          // 1572864
#define OUT_PROB  (OUT_IDX + NTOK * 2)     // 1576960
#define OUT_LOSS  (OUT_PROB + NTOK * 2)    // 1581056
#define OUT_CNT   (OUT_LOSS + 1)           // 1581057

typedef __attribute__((ext_vector_type(16))) __bf16        v16bf;
typedef __attribute__((ext_vector_type(2)))  __bf16        v2bf;
typedef __attribute__((ext_vector_type(8)))  float         v8f;
typedef __attribute__((ext_vector_type(8)))  unsigned int  v8u;

__device__ __forceinline__ unsigned short f2bf(float f) {
  unsigned int u = __float_as_uint(f);
  u += 0x7FFFu + ((u >> 16) & 1u);            // round-to-nearest-even
  return (unsigned short)(u >> 16);
}
__device__ __forceinline__ unsigned int cvt_pk_bf16(float f0, float f1) {
#if __has_builtin(__builtin_amdgcn_cvt_pk_bf16_f32)
  v2bf r = __builtin_amdgcn_cvt_pk_bf16_f32(f0, f1);
  return __builtin_bit_cast(unsigned int, r);
#else
  return (unsigned int)f2bf(f0) | ((unsigned int)f2bf(f1) << 16);
#endif
}
__device__ __forceinline__ float silu(float v) {
  return v * __builtin_amdgcn_rcpf(1.0f + __expf(-v));
}
__device__ __forceinline__ v16bf frag_cast(uint4 lo, uint4 hi) {
  v8u u; u[0]=lo.x; u[1]=lo.y; u[2]=lo.z; u[3]=lo.w;
         u[4]=hi.x; u[5]=hi.y; u[6]=hi.z; u[7]=hi.w;
  return __builtin_bit_cast(v16bf, u);
}

// ---------------- kernel 1: reset counters ----------------
__global__ void moe_zero_cnt(int* __restrict__ wsI) {
  if (threadIdx.x < NGROUP) wsI[threadIdx.x] = 0;
}

// ---------------- kernel: pack weights [E][K][N] f32 -> B-fragment bf16 layout ----------------
// one fragment = 32 lanes x 8 uints; lane = h*16+nl supplies K = kb+h*8+2(v&3)+16(v>>2)(+1), N = nt*16+nl
__global__ void __launch_bounds__(256)
moe_pack_b(const float* __restrict__ src, unsigned int* __restrict__ dst, int K, int N) {
  const int nt  = blockIdx.x;
  const int e   = blockIdx.y;
  const int tid = threadIdx.x;
  const int lane = tid & 31, v = tid >> 5;
  const int h = lane >> 4, nl = lane & 15;
  const float* s = src + (size_t)e * K * N + (size_t)nt * 16 + nl;
  unsigned int* d = dst + (((size_t)e * (N / 16) + nt) * (size_t)(K / 32)) * 256 + lane * 8 + v;
  const int nkb = K / 32;
  for (int kbi = 0; kbi < nkb; ++kbi) {
    const int k0 = kbi * 32 + h * 8 + 2 * (v & 3) + 16 * (v >> 2);
    const float f0 = s[(size_t)k0 * N];
    const float f1 = s[(size_t)(k0 + 1) * N];
    d[(size_t)kbi * 256] = cvt_pk_bf16(f0, f1);
  }
}

// ---------------- kernel 2: router (gate + softmax + top2) ----------------
__global__ void __launch_bounds__(256)
moe_router(const float* __restrict__ x, const float* __restrict__ gw,
           float* __restrict__ out, int* __restrict__ wsI, float* __restrict__ wsF) {
  __shared__ float sg[HIDDEN * NE];                       // 24 KB
  for (int i = threadIdx.x; i < HIDDEN * NE; i += 256) sg[i] = gw[i];
  __syncthreads();

  const int t = blockIdx.x * 256 + threadIdx.x;
  if (t >= NTOK) return;
  const float* xr = x + (size_t)t * HIDDEN;

  float acc[NE];
#pragma unroll
  for (int e = 0; e < NE; ++e) acc[e] = 0.0f;
  for (int k = 0; k < HIDDEN; ++k) {
    const float xv = xr[k];
#pragma unroll
    for (int e = 0; e < NE; ++e) acc[e] = fmaf(xv, sg[k * NE + e], acc[e]);
  }
  float m = acc[0];
#pragma unroll
  for (int e = 1; e < NE; ++e) m = fmaxf(m, acc[e]);
  float s = 0.0f;
#pragma unroll
  for (int e = 0; e < NE; ++e) { acc[e] = __expf(acc[e] - m); s += acc[e]; }
  const float inv = __builtin_amdgcn_rcpf(s);
#pragma unroll
  for (int e = 0; e < NE; ++e) acc[e] *= inv;
  int i0 = 0; float p0 = acc[0];
#pragma unroll
  for (int e = 1; e < NE; ++e) if (acc[e] > p0) { p0 = acc[e]; i0 = e; }
  int i1 = (i0 == 0) ? 1 : 0; float p1 = acc[i1];
#pragma unroll
  for (int e = 0; e < NE; ++e) if (e != i0 && acc[e] > p1) { p1 = acc[e]; i1 = e; }

  out[OUT_IDX + 2 * t]      = (float)i0;
  out[OUT_IDX + 2 * t + 1]  = (float)i1;
  out[OUT_PROB + 2 * t]     = p0;
  out[OUT_PROB + 2 * t + 1] = p1;

  const int g0 = i0 * 2, g1 = i1 * 2 + 1;
  const int s0 = atomicAdd(&wsI[g0], 1);
  wsI[OFF_TOK + g0 * CAP + s0] = t;
  wsF[OFF_WT  + g0 * CAP + s0] = p0;
  const int s1 = atomicAdd(&wsI[g1], 1);
  wsI[OFF_TOK + g1 * CAP + s1] = t;
  wsF[OFF_WT  + g1 * CAP + s1] = p1;
}

// ---------------- kernel 3: prefix sum + aux outputs ----------------
__global__ void moe_finalize(int* __restrict__ wsI, float* __restrict__ out) {
  if (threadIdx.x != 0) return;
  int b = 0;
  for (int g = 0; g < NGROUP; ++g) { wsI[16 + g] = b; b += wsI[g]; }
  float l = 0.0f;
  for (int e = 0; e < NE; ++e) {
    const float c = (float)(wsI[2 * e] + wsI[2 * e + 1]);
    out[OUT_CNT + e] = c;
    const float f = c / (float)(NTOK * 2);
    l += f * f;
  }
  out[OUT_LOSS] = (float)NE * l;
}

// ---------------- kernel 4: h1 = silu(x @ W1[e])  (grouped, bf16 WMMA) ----------------
// block = 128 threads (4 waves), tile = 16 tokens x 128 inter cols; wave tile 16x32.
__global__ void __launch_bounds__(128)
moe_ffn1(const float* __restrict__ x, const unsigned int* __restrict__ w1p,
         const int* __restrict__ wsI, unsigned short* __restrict__ h1) {
  const int g  = blockIdx.z;             // expert*2 + kslot
  const int e  = g >> 1;
  const int Ce = wsI[g];
  const int t0 = blockIdx.y * 16;
  if (t0 >= Ce) return;
  const int gb   = wsI[16 + g];
  const int tid  = threadIdx.x;
  const int wave = tid >> 5, lane = tid & 31;
  const int h    = lane >> 4, nl = lane & 15;
  const int n0   = blockIdx.x * 128 + wave * 32;

  // stage gathered x rows into LDS as bf16 (shared by all 4 waves)
  __shared__ __align__(16) unsigned short sA[16 * HIDDEN];   // 24 KB
  unsigned int* sA32 = (unsigned int*)sA;
  for (int i = tid; i < 16 * (HIDDEN / 2); i += 128) {
    const int row = i / (HIDDEN / 2), kp = i - row * (HIDDEN / 2);
    unsigned int v = 0;
    const int slot = t0 + row;
    if (slot < Ce) {
      const int tok = wsI[OFF_TOK + g * CAP + slot];
      const float* xp = x + (size_t)tok * HIDDEN + 2 * kp;
      v = cvt_pk_bf16(xp[0], xp[1]);
    }
    sA32[i] = v;
  }
  __syncthreads();

  const unsigned short* arow = &sA[(lane & 15) * HIDDEN];    // A: M = lane%16
  const size_t ntStride = (size_t)(HIDDEN / 32) * 256;       // uints between n-tiles
  const unsigned int* bp =
      w1p + (((size_t)e * (INTER / 16) + (n0 >> 4)) * (size_t)(HIDDEN / 32)) * 256 + lane * 8;
  v8f c0 = {}, c1 = {};
#pragma unroll 2
  for (int kbi = 0; kbi < HIDDEN / 32; ++kbi) {
    const int kb = kbi * 32;
    const v16bf a = frag_cast(*(const uint4*)(arow + kb + h * 8),
                              *(const uint4*)(arow + kb + 16 + h * 8));
    const unsigned int* b = bp + (size_t)kbi * 256;
    const v16bf b0 = frag_cast(*(const uint4*)(b),            *(const uint4*)(b + 4));
    const v16bf b1 = frag_cast(*(const uint4*)(b + ntStride), *(const uint4*)(b + ntStride + 4));
    c0 = __builtin_amdgcn_wmma_f32_16x16x32_bf16(false, a, false, b0, (short)0, c0, false, false);
    c1 = __builtin_amdgcn_wmma_f32_16x16x32_bf16(false, a, false, b1, (short)0, c1, false, false);
  }
  // D layout: VGPR r -> M = r + 8*h, N = lane%16 ; SiLU, store bf16 h1
#pragma unroll
  for (int r = 0; r < 8; ++r) {
    const int slot = t0 + r + h * 8;
    if (slot < Ce) {
      unsigned short* hp = h1 + (size_t)(gb + slot) * INTER + n0 + nl;
      hp[0]  = f2bf(silu(c0[r]));
      hp[16] = f2bf(silu(c1[r]));
    }
  }
}

// ---------------- kernel 5: y (+)= combine * (h1 @ W2[e]) ----------------
__global__ void __launch_bounds__(128)
moe_ffn2(const unsigned short* __restrict__ h1, const unsigned int* __restrict__ w2p,
         const int* __restrict__ wsI, const float* __restrict__ wsF,
         float* __restrict__ y, int jsel) {
  const int e  = blockIdx.z;
  const int g  = e * 2 + jsel;
  const int Ce = wsI[g];
  const int t0 = blockIdx.y * 16;
  if (t0 >= Ce) return;
  const int gb   = wsI[16 + g];
  const int tid  = threadIdx.x;
  const int wave = tid >> 5, lane = tid & 31;
  const int h    = lane >> 4, nl = lane & 15;
  const int n0   = blockIdx.x * 128 + wave * 32;

  int ar = t0 + (lane & 15);
  if (ar >= Ce) ar = Ce - 1;                    // clamp padded rows (stay in-bounds)
  const unsigned short* arow = h1 + (size_t)(gb + ar) * INTER;
  const size_t ntStride = (size_t)(INTER / 32) * 256;
  const unsigned int* bp =
      w2p + (((size_t)e * (HIDDEN / 16) + (n0 >> 4)) * (size_t)(INTER / 32)) * 256 + lane * 8;
  v8f c0 = {}, c1 = {};
#pragma unroll 2
  for (int kbi = 0; kbi < INTER / 32; ++kbi) {
    const int kb = kbi * 32;
    const v16bf a = frag_cast(*(const uint4*)(arow + kb + h * 8),
                              *(const uint4*)(arow + kb + 16 + h * 8));
    const unsigned int* b = bp + (size_t)kbi * 256;
    const v16bf b0 = frag_cast(*(const uint4*)(b),            *(const uint4*)(b + 4));
    const v16bf b1 = frag_cast(*(const uint4*)(b + ntStride), *(const uint4*)(b + ntStride + 4));
    c0 = __builtin_amdgcn_wmma_f32_16x16x32_bf16(false, a, false, b0, (short)0, c0, false, false);
    c1 = __builtin_amdgcn_wmma_f32_16x16x32_bf16(false, a, false, b1, (short)0, c1, false, false);
  }
#pragma unroll
  for (int r = 0; r < 8; ++r) {
    const int slot = t0 + r + h * 8;
    if (slot < Ce) {
      const int   tok = wsI[OFF_TOK + g * CAP + slot];
      const float wt  = wsF[OFF_WT  + g * CAP + slot];
      float* yp = y + (size_t)tok * HIDDEN + n0 + nl;
      const float v0 = wt * c0[r];
      const float v1 = wt * c1[r];
      if (jsel) { yp[0] += v0; yp[16] += v1; }  // pass 2: exclusive per token
      else      { yp[0]  = v0; yp[16]  = v1; }  // pass 1: initializes y
    }
  }
}

extern "C" void kernel_launch(void* const* d_in, const int* in_sizes, int n_in,
                              void* d_out, int out_size, void* d_ws, size_t ws_size,
                              hipStream_t stream) {
  const float* x  = (const float*)d_in[0];
  const float* gw = (const float*)d_in[1];
  const float* w1 = (const float*)d_in[2];
  const float* w2 = (const float*)d_in[3];
  float* out = (float*)d_out;
  int*   wsI = (int*)d_ws;
  float* wsF = (float*)d_ws;
  unsigned short* h1  = (unsigned short*)((char*)d_ws + OFF_H1_BYTES);
  unsigned int*   w1p = (unsigned int*)((char*)d_ws + OFF_W1P_BYTES);
  unsigned int*   w2p = (unsigned int*)((char*)d_ws + OFF_W2P_BYTES);

  moe_zero_cnt<<<1, 32, 0, stream>>>(wsI);
  // pack weights into WMMA B-fragment layout (streaming, once per element)
  moe_pack_b<<<dim3(INTER / 16, NE), 256, 0, stream>>>(w1, w1p, HIDDEN, INTER);
  moe_pack_b<<<dim3(HIDDEN / 16, NE), 256, 0, stream>>>(w2, w2p, INTER, HIDDEN);

  moe_router<<<NTOK / 256, 256, 0, stream>>>(x, gw, out, wsI, wsF);
  moe_finalize<<<1, 32, 0, stream>>>(wsI, out);

  dim3 gA(INTER / 128, NTOK / 16, NGROUP);
  moe_ffn1<<<gA, 128, 0, stream>>>(x, w1p, wsI, h1);

  dim3 gB(HIDDEN / 128, NTOK / 16, NE);
  moe_ffn2<<<gB, 128, 0, stream>>>(h1, w2p, wsI, wsF, out, 0);  // top-1 pass: y =
  moe_ffn2<<<gB, 128, 0, stream>>>(h1, w2p, wsI, wsF, out, 1);  // top-2 pass: y +=
}
